// LidarCameraFusionMambaV2_16449724745542
// MI455X (gfx1250) — matrix operand order repaired and checked by reading, not compiled, not run on hardware
//
#include <hip/hip_runtime.h>
#include <hip/hip_bf16.h>
#include <math.h>

// ---- problem dimensions (fixed by the reference) ----
#define DMODEL 512
#define DSTATE 16
#define DCONV  4
#define DINNER 1024
#define DTRANK 32
#define BB     2
#define LSEQ   2048
#define MROWS  (BB * LSEQ)            // 4096 token rows
#define SSMW   (DTRANK + 2 * DSTATE)  // 64
#define NCH    8                      // scan chunks per sequence
#define CH     (LSEQ / NCH)           // 256 steps per chunk

typedef __bf16 bf16_t;
typedef __attribute__((ext_vector_type(16))) __bf16 bf16x16;
typedef __attribute__((ext_vector_type(8)))  __bf16 bf16x8;
typedef __attribute__((ext_vector_type(8)))  float  f32x8;

// ---------------- helpers ----------------
__device__ __forceinline__ bf16_t to_bf16(float f) {
  unsigned u = __float_as_uint(f);
  unsigned r = ((u >> 16) & 1u) + 0x7fffu;           // round-to-nearest-even
  unsigned short h = (unsigned short)((u + r) >> 16);
  union { unsigned short s; bf16_t b; } cv; cv.s = h;
  return cv.b;
}
__device__ __forceinline__ float sigmoidf_(float x) {
  return 1.0f / (1.0f + __expf(-x));
}

// gfx1250 async global->LDS copy (ASYNCcnt tracked), 16 bytes per lane
__device__ __forceinline__ void async_ld_b128(unsigned lds_off, const void* gaddr) {
  asm volatile("global_load_async_to_lds_b128 %0, %1, off"
               :: "v"(lds_off), "v"(gaddr) : "memory");
}
__device__ __forceinline__ void wait_async0() {
  asm volatile("s_wait_asynccnt 0" ::: "memory");
}

// ---------------- WMMA fragments (CDNA5 16x16x32 bf16 layout) ----------------
// A tile 16x32: lane<16 -> row=lane, K chunks {k..k+7},{k+16..k+23}
//               lane>=16 -> row=lane-16, K chunks {k+8..k+15},{k+24..k+31}
__device__ __forceinline__ bf16x16 frag_a(const bf16_t* __restrict__ A, int lda,
                                          int m0, int k0, int lane) {
  const bf16_t* p = A + (size_t)(m0 + (lane & 15)) * lda + k0 + ((lane >> 4) << 3);
  bf16x8 lo = *(const bf16x8*)p;
  bf16x8 hi = *(const bf16x8*)(p + 16);
  bf16x16 r;
#pragma unroll
  for (int i = 0; i < 8; ++i) { r[i] = lo[i]; r[i + 8] = hi[i]; }
  return r;
}
// B tile 32x16 read from LDS-staged W rows (row stride 64B = 32 bf16):
// lane<16 -> col n_local+lane, K {0..7},{16..23}; lane>=16 -> K {8..15},{24..31}
__device__ __forceinline__ bf16x16 frag_b_lds(const unsigned char* sbase, int n_local, int lane) {
  const unsigned char* p = sbase + (size_t)(n_local + (lane & 15)) * 64 + ((lane >> 4) << 4);
  bf16x8 lo = *(const bf16x8*)p;
  bf16x8 hi = *(const bf16x8*)(p + 32);
  bf16x16 r;
#pragma unroll
  for (int i = 0; i < 8; ++i) { r[i] = lo[i]; r[i + 8] = hi[i]; }
  return r;
}
// C/D 16x16 f32: VGPR i -> row m0+i (lanes 0..15) / m0+8+i (lanes 16..31), col n0+(lane&15)
__device__ __forceinline__ void store_c(float* __restrict__ C, int ldc,
                                        int m0, int n0, int lane, f32x8 acc) {
  float* p = C + (size_t)(m0 + ((lane >> 4) << 3)) * ldc + n0 + (lane & 15);
#pragma unroll
  for (int i = 0; i < 8; ++i) p[(size_t)i * ldc] = acc[i];
}

// ---------------- GEMM: C[M,N] = A[M,K] * W[N,K]^T, N multiple of 64 ----------------
// block = 256 threads = 8 waves; block tile 128x64; W tile (64x32 bf16 = 4KB)
// double-buffered in LDS via async-to-LDS; grid must tile M,N exactly.
__global__ void gemm_wmma_n64(const bf16_t* __restrict__ A, const bf16_t* __restrict__ W,
                              float* __restrict__ C, int M, int N, int K,
                              int lda, int ldw, int ldc) {
  __shared__ alignas(16) unsigned char sW[2][64 * 64];
  const int tid  = threadIdx.x;
  const int lane = tid & 31;
  const int wave = tid >> 5;
  const int m0 = (blockIdx.y * 8 + wave) * 16;
  const int n0 = blockIdx.x * 64;

  const int srow = tid >> 2;        // 0..63
  const int sseg = tid & 3;         // 16B segment within 64B row
  const unsigned loff0 = (unsigned)(size_t)(&sW[0][srow * 64 + sseg * 16]);
  const unsigned loff1 = (unsigned)(size_t)(&sW[1][srow * 64 + sseg * 16]);
  const bf16_t* gW = W + (size_t)(n0 + srow) * ldw + sseg * 8;

  f32x8 acc0 = {}, acc1 = {}, acc2 = {}, acc3 = {};
  async_ld_b128(loff0, (const void*)gW);
  wait_async0();
  __syncthreads();
  for (int k = 0; k < K; k += 32) {
    const int cur = (k >> 5) & 1;
    if (k + 32 < K)  // stream next W tile into the other buffer while computing
      async_ld_b128(cur ? loff0 : loff1, (const void*)(gW + k + 32));
    if (k + 32 < K)
      __builtin_prefetch((const void*)(A + (size_t)(m0 + (lane & 15)) * lda + k + 32), 0, 1);
    bf16x16 a  = frag_a(A, lda, m0, k, lane);
    const unsigned char* sb = &sW[cur][0];
    bf16x16 b0 = frag_b_lds(sb,  0, lane);
    bf16x16 b1 = frag_b_lds(sb, 16, lane);
    bf16x16 b2 = frag_b_lds(sb, 32, lane);
    bf16x16 b3 = frag_b_lds(sb, 48, lane);
    acc0 = __builtin_amdgcn_wmma_f32_16x16x32_bf16(false, a, false, b0, (short)0, acc0, false, false);
    acc1 = __builtin_amdgcn_wmma_f32_16x16x32_bf16(false, a, false, b1, (short)0, acc1, false, false);
    acc2 = __builtin_amdgcn_wmma_f32_16x16x32_bf16(false, a, false, b2, (short)0, acc2, false, false);
    acc3 = __builtin_amdgcn_wmma_f32_16x16x32_bf16(false, a, false, b3, (short)0, acc3, false, false);
    wait_async0();
    __syncthreads();
  }
  store_c(C, ldc, m0, n0 +  0, lane, acc0);
  store_c(C, ldc, m0, n0 + 16, lane, acc1);
  store_c(C, ldc, m0, n0 + 32, lane, acc2);
  store_c(C, ldc, m0, n0 + 48, lane, acc3);
}

// ---------------- GEMM variant: N multiple of 16 (used for N=64 ssm projection) ----------------
__global__ void gemm_wmma_n16(const bf16_t* __restrict__ A, const bf16_t* __restrict__ W,
                              float* __restrict__ C, int M, int N, int K,
                              int lda, int ldw, int ldc) {
  __shared__ alignas(16) unsigned char sW[2][16 * 64];
  const int tid  = threadIdx.x;
  const int lane = tid & 31;
  const int wave = tid >> 5;
  const int m0 = (blockIdx.y * 8 + wave) * 16;
  const int n0 = blockIdx.x * 16;

  const int srow = tid >> 2;   // rows 0..15 handled by tids 0..63
  const int sseg = tid & 3;
  const bool stager = (tid < 64);
  const unsigned loff0 = (unsigned)(size_t)(&sW[0][srow * 64 + sseg * 16]);
  const unsigned loff1 = (unsigned)(size_t)(&sW[1][srow * 64 + sseg * 16]);
  const bf16_t* gW = W + (size_t)(n0 + srow) * ldw + sseg * 8;

  f32x8 acc = {};
  if (stager) async_ld_b128(loff0, (const void*)gW);
  wait_async0();
  __syncthreads();
  for (int k = 0; k < K; k += 32) {
    const int cur = (k >> 5) & 1;
    if (stager && (k + 32 < K))
      async_ld_b128(cur ? loff0 : loff1, (const void*)(gW + k + 32));
    bf16x16 a = frag_a(A, lda, m0, k, lane);
    bf16x16 b = frag_b_lds(&sW[cur][0], 0, lane);
    acc = __builtin_amdgcn_wmma_f32_16x16x32_bf16(false, a, false, b, (short)0, acc, false, false);
    wait_async0();
    __syncthreads();
  }
  store_c(C, ldc, m0, n0, lane, acc);
}

// ---------------- f32 -> bf16 (optionally strided source) ----------------
__global__ void cvt_bf16(const float* __restrict__ src, bf16_t* __restrict__ dst,
                         int rows, int width, int srcld) {
  int i = blockIdx.x * blockDim.x + threadIdx.x;
  int n = rows * width;
  if (i >= n) return;
  int r = i / width, c = i - r * width;
  dst[i] = to_bf16(src[(size_t)r * srcld + c]);
}

// ---------------- gate: cam_xg = cam_x * sigmoid(glin + lidar_b); also cam_x -> bf16 ----------------
__global__ void gate_fuse(const float* __restrict__ cam_xz, const float* __restrict__ glin,
                          const float* __restrict__ lidar_b,
                          float* __restrict__ cam_xg, bf16_t* __restrict__ cam_x_bf) {
  int i = blockIdx.x * blockDim.x + threadIdx.x;
  if (i >= MROWS * DINNER) return;
  int d = i & (DINNER - 1);
  int r = i >> 10;
  float cx = cam_xz[(size_t)r * (2 * DINNER) + d];
  float g  = sigmoidf_(glin[i] + lidar_b[d]);
  cam_xg[i]   = cx * g;
  cam_x_bf[i] = to_bf16(cx);
}

// ---------------- causal depthwise conv (4 taps) + SiLU; dir=1 means time-reversed ----------------
__global__ void conv_silu(const float* __restrict__ x, const float* __restrict__ w4,
                          const float* __restrict__ bias, float* __restrict__ out, int dir) {
  int i = blockIdx.x * blockDim.x + threadIdx.x;
  if (i >= MROWS * DINNER) return;
  int d = i & (DINNER - 1);
  int r = i >> 10;
  int l = r & (LSEQ - 1);
  float acc = bias[d];
  const float* w = w4 + (size_t)d * DCONV;
  if (!dir) {
#pragma unroll
    for (int j = 0; j < DCONV; ++j) {       // out[l] = sum_j w[j] * x[l-3+j]
      int ls = l - (DCONV - 1) + j;
      if (ls >= 0) acc += w[j] * x[(size_t)(r - (DCONV - 1) + j) * DINNER + d];
    }
  } else {
#pragma unroll
    for (int j = 0; j < DCONV; ++j) {       // flipped sequence, stored in natural order
      int ls = l + (DCONV - 1) - j;
      if (ls < LSEQ) acc += w[j] * x[(size_t)(r + (DCONV - 1) - j) * DINNER + d];
    }
  }
  out[i] = acc * sigmoidf_(acc);
}

// ---------------- chunked selective scan ----------------
// recurrence per (b,d,n): h <- a*h + dt*u*B_n, a = exp(dt*A_n), dt = softplus(delta + bias)
// phase A: per chunk, P = prod(a), S = h_end with h_in = 0
__global__ void scan_partial(const float* __restrict__ delta, const float* __restrict__ ssm_p,
                             const float* __restrict__ u_act, const float* __restrict__ A_log,
                             const float* __restrict__ dt_b,
                             float* __restrict__ Pbuf, float* __restrict__ Sbuf, int dir) {
  int tid = blockIdx.x * blockDim.x + threadIdx.x;
  if (tid >= BB * DINNER * NCH) return;
  int d = tid & (DINNER - 1);
  int c = (tid >> 10) & (NCH - 1);
  int b = tid >> 13;
  float An[DSTATE], pa[DSTATE], h[DSTATE];
#pragma unroll
  for (int n = 0; n < DSTATE; ++n) {
    An[n] = -__expf(A_log[(size_t)d * DSTATE + n]);
    pa[n] = 1.0f; h[n] = 0.0f;
  }
  const float bia = dt_b[d];
  for (int s = 0; s < CH; ++s) {
    int t = c * CH + s;                       // logical time
    int l = dir ? (LSEQ - 1 - t) : t;         // natural index
    size_t r = (size_t)b * LSEQ + l;
    float dt = delta[r * DINNER + d] + bia;
    dt = (dt > 20.0f) ? dt : log1pf(__expf(dt));
    float du = dt * u_act[r * DINNER + d];
    const float* sp = ssm_p + r * SSMW;
#pragma unroll
    for (int n = 0; n < DSTATE; ++n) {
      float a = __expf(dt * An[n]);
      pa[n] *= a;
      h[n] = a * h[n] + du * sp[DTRANK + n];
    }
  }
  size_t base = (((size_t)b * DINNER + d) * NCH + c) * DSTATE;
#pragma unroll
  for (int n = 0; n < DSTATE; ++n) { Pbuf[base + n] = pa[n]; Sbuf[base + n] = h[n]; }
}

// phase B: 8-step sequential carry per (b,d); Hin[c] = state entering chunk c
__global__ void scan_carry(const float* __restrict__ Pbuf, const float* __restrict__ Sbuf,
                           float* __restrict__ Hin) {
  int tid = blockIdx.x * blockDim.x + threadIdx.x;
  if (tid >= BB * DINNER) return;
  size_t base = (size_t)tid * NCH * DSTATE;
  float h[DSTATE];
#pragma unroll
  for (int n = 0; n < DSTATE; ++n) h[n] = 0.0f;
  for (int c = 0; c < NCH; ++c) {
    size_t o = base + (size_t)c * DSTATE;
#pragma unroll
    for (int n = 0; n < DSTATE; ++n) {
      Hin[o + n] = h[n];
      h[n] = Pbuf[o + n] * h[n] + Sbuf[o + n];
    }
  }
}

// phase C: re-run chunk with correct carry-in, emit y = (C.h + D*u) * silu(z)
__global__ void scan_apply(const float* __restrict__ delta, const float* __restrict__ ssm_p,
                           const float* __restrict__ u_act, const float* __restrict__ cam_xz,
                           const float* __restrict__ A_log, const float* __restrict__ Dw,
                           const float* __restrict__ dt_b, const float* __restrict__ Hin,
                           float* __restrict__ y, int dir) {
  int tid = blockIdx.x * blockDim.x + threadIdx.x;
  if (tid >= BB * DINNER * NCH) return;
  int d = tid & (DINNER - 1);
  int c = (tid >> 10) & (NCH - 1);
  int b = tid >> 13;
  float An[DSTATE], h[DSTATE];
  size_t base = (((size_t)b * DINNER + d) * NCH + c) * DSTATE;
#pragma unroll
  for (int n = 0; n < DSTATE; ++n) {
    An[n] = -__expf(A_log[(size_t)d * DSTATE + n]);
    h[n] = Hin[base + n];
  }
  const float Dd  = Dw[d];
  const float bia = dt_b[d];
  for (int s = 0; s < CH; ++s) {
    int t = c * CH + s;
    int l = dir ? (LSEQ - 1 - t) : t;
    size_t r = (size_t)b * LSEQ + l;
    float dt = delta[r * DINNER + d] + bia;
    dt = (dt > 20.0f) ? dt : log1pf(__expf(dt));
    float uu = u_act[r * DINNER + d];
    float du = dt * uu;
    const float* sp = ssm_p + r * SSMW;
    float yy = 0.0f;
#pragma unroll
    for (int n = 0; n < DSTATE; ++n) {
      float a = __expf(dt * An[n]);
      h[n] = a * h[n] + du * sp[DTRANK + n];
      yy += h[n] * sp[DTRANK + DSTATE + n];
    }
    float zz = cam_xz[r * (size_t)(2 * DINNER) + DINNER + d];
    y[r * DINNER + d] = (yy + Dd * uu) * (zz * sigmoidf_(zz));
  }
}

// ---------------- y_h2t + y_t2h -> bf16 ----------------
__global__ void combine_bf16(const float* __restrict__ a, const float* __restrict__ b,
                             bf16_t* __restrict__ dst, int n) {
  int i = blockIdx.x * blockDim.x + threadIdx.x;
  if (i >= n) return;
  dst[i] = to_bf16(a[i] + b[i]);
}

// ---------------- host side ----------------
static inline void* wsal(char*& p, size_t bytes) {
  void* r = (void*)p;
  p += (bytes + 255) & ~(size_t)255;
  return r;
}
static inline int cdiv(int a, int b) { return (a + b - 1) / b; }

extern "C" void kernel_launch(void* const* d_in, const int* in_sizes, int n_in,
                              void* d_out, int out_size, void* d_ws, size_t ws_size,
                              hipStream_t stream) {
  (void)in_sizes; (void)n_in; (void)out_size; (void)ws_size;
  const float* lidar_feats = (const float*)d_in[0];
  const float* camera_feats= (const float*)d_in[1];
  const float* in_proj_w   = (const float*)d_in[2];
  const float* conv_w_h2t  = (const float*)d_in[3];
  const float* conv_b_h2t  = (const float*)d_in[4];
  const float* x_proj_w_h2t= (const float*)d_in[5];
  const float* dt_w_h2t    = (const float*)d_in[6];
  const float* dt_b_h2t    = (const float*)d_in[7];
  const float* A_log_h2t   = (const float*)d_in[8];
  const float* D_h2t       = (const float*)d_in[9];
  const float* conv_w_t2h  = (const float*)d_in[10];
  const float* conv_b_t2h  = (const float*)d_in[11];
  const float* x_proj_w_t2h= (const float*)d_in[12];
  const float* dt_w_t2h    = (const float*)d_in[13];
  const float* dt_b_t2h    = (const float*)d_in[14];
  const float* A_log_t2h   = (const float*)d_in[15];
  const float* D_t2h       = (const float*)d_in[16];
  const float* lidar_w     = (const float*)d_in[17];
  const float* lidar_b     = (const float*)d_in[18];
  const float* out_proj_w  = (const float*)d_in[19];
  float* out = (float*)d_out;

  char* wp = (char*)d_ws;
  // f32 scratch
  float* cam_xz  = (float*)wsal(wp, (size_t)MROWS * 2 * DINNER * 4);
  float* lidar_x = (float*)wsal(wp, (size_t)MROWS * DINNER * 4);
  float* glin    = (float*)wsal(wp, (size_t)MROWS * DINNER * 4);
  float* cam_xg  = (float*)wsal(wp, (size_t)MROWS * DINNER * 4);
  float* ssm_p0  = (float*)wsal(wp, (size_t)MROWS * SSMW * 4);
  float* ssm_p1  = (float*)wsal(wp, (size_t)MROWS * SSMW * 4);
  float* delta0  = (float*)wsal(wp, (size_t)MROWS * DINNER * 4);
  float* delta1  = (float*)wsal(wp, (size_t)MROWS * DINNER * 4);
  float* xact0   = (float*)wsal(wp, (size_t)MROWS * DINNER * 4);
  float* xact1   = (float*)wsal(wp, (size_t)MROWS * DINNER * 4);
  float* y0      = (float*)wsal(wp, (size_t)MROWS * DINNER * 4);
  float* y1      = (float*)wsal(wp, (size_t)MROWS * DINNER * 4);
  // chunked-scan scratch (per direction): P, S, Hin  [B][DINNER][NCH][DSTATE]
  size_t scanSz = (size_t)BB * DINNER * NCH * DSTATE * 4;
  float* P0 = (float*)wsal(wp, scanSz);
  float* S0 = (float*)wsal(wp, scanSz);
  float* H0 = (float*)wsal(wp, scanSz);
  float* P1 = (float*)wsal(wp, scanSz);
  float* S1 = (float*)wsal(wp, scanSz);
  float* H1 = (float*)wsal(wp, scanSz);
  // bf16 scratch
  bf16_t* cam_bf   = (bf16_t*)wsal(wp, (size_t)MROWS * DMODEL * 2);
  bf16_t* lid_bf   = (bf16_t*)wsal(wp, (size_t)MROWS * DMODEL * 2);
  bf16_t* w_in_bf  = (bf16_t*)wsal(wp, (size_t)2 * DINNER * DMODEL * 2);
  bf16_t* w_gate_bf= (bf16_t*)wsal(wp, (size_t)DINNER * DINNER * 2);
  bf16_t* w_out_bf = (bf16_t*)wsal(wp, (size_t)DMODEL * DINNER * 2);
  bf16_t* w_xp0_bf = (bf16_t*)wsal(wp, (size_t)SSMW * DINNER * 2);
  bf16_t* w_xp1_bf = (bf16_t*)wsal(wp, (size_t)SSMW * DINNER * 2);
  bf16_t* w_dt0_bf = (bf16_t*)wsal(wp, (size_t)DINNER * DTRANK * 2);
  bf16_t* w_dt1_bf = (bf16_t*)wsal(wp, (size_t)DINNER * DTRANK * 2);
  bf16_t* cam_x_bf = (bf16_t*)wsal(wp, (size_t)MROWS * DINNER * 2);
  bf16_t* lidx_bf  = (bf16_t*)wsal(wp, (size_t)MROWS * DINNER * 2);
  bf16_t* dtr0_bf  = (bf16_t*)wsal(wp, (size_t)MROWS * DTRANK * 2);
  bf16_t* dtr1_bf  = (bf16_t*)wsal(wp, (size_t)MROWS * DTRANK * 2);
  bf16_t* ysum_bf  = (bf16_t*)wsal(wp, (size_t)MROWS * DINNER * 2);

  const int TB = 256;
#define CVT(src, dst, rows, width, srcld) \
  cvt_bf16<<<cdiv((rows) * (width), TB), TB, 0, stream>>>((src), (dst), (rows), (width), (srcld))

  // 1) bf16 staging of activations and weights
  CVT(camera_feats, cam_bf, MROWS, DMODEL, DMODEL);
  CVT(lidar_feats,  lid_bf, MROWS, DMODEL, DMODEL);
  CVT(in_proj_w,    w_in_bf, 2 * DINNER, DMODEL, DMODEL);
  CVT(lidar_w,      w_gate_bf, DINNER, DINNER, DINNER);
  CVT(out_proj_w,   w_out_bf, DMODEL, DINNER, DINNER);
  CVT(x_proj_w_h2t, w_xp0_bf, SSMW, DINNER, DINNER);
  CVT(x_proj_w_t2h, w_xp1_bf, SSMW, DINNER, DINNER);
  CVT(dt_w_h2t,     w_dt0_bf, DINNER, DTRANK, DTRANK);
  CVT(dt_w_t2h,     w_dt1_bf, DINNER, DTRANK, DTRANK);

  dim3 blk(256);
  // 2) cam_xz = camera @ in_proj_w^T   [4096 x 2048, K=512]
  gemm_wmma_n64<<<dim3(2 * DINNER / 64, MROWS / 128), blk, 0, stream>>>(
      cam_bf, w_in_bf, cam_xz, MROWS, 2 * DINNER, DMODEL, DMODEL, DMODEL, 2 * DINNER);
  // 3) lidar_x = lidar @ in_proj_w[:1024]^T   [4096 x 1024, K=512]
  gemm_wmma_n64<<<dim3(DINNER / 64, MROWS / 128), blk, 0, stream>>>(
      lid_bf, w_in_bf, lidar_x, MROWS, DINNER, DMODEL, DMODEL, DMODEL, DINNER);
  CVT(lidar_x, lidx_bf, MROWS, DINNER, DINNER);
  // 4) glin = lidar_x @ lidar_w^T   [4096 x 1024, K=1024]
  gemm_wmma_n64<<<dim3(DINNER / 64, MROWS / 128), blk, 0, stream>>>(
      lidx_bf, w_gate_bf, glin, MROWS, DINNER, DINNER, DINNER, DINNER, DINNER);
  // 5) gate fuse
  gate_fuse<<<cdiv(MROWS * DINNER, TB), TB, 0, stream>>>(cam_xz, glin, lidar_b, cam_xg, cam_x_bf);
  // 6) ssm projections: cam_x @ x_proj^T   [4096 x 64, K=1024]
  gemm_wmma_n16<<<dim3(SSMW / 16, MROWS / 128), blk, 0, stream>>>(
      cam_x_bf, w_xp0_bf, ssm_p0, MROWS, SSMW, DINNER, DINNER, DINNER, SSMW);
  gemm_wmma_n16<<<dim3(SSMW / 16, MROWS / 128), blk, 0, stream>>>(
      cam_x_bf, w_xp1_bf, ssm_p1, MROWS, SSMW, DINNER, DINNER, DINNER, SSMW);
  // 7) dt_r slices -> bf16, delta = dt_r @ dt_w^T   [4096 x 1024, K=32]
  CVT(ssm_p0, dtr0_bf, MROWS, DTRANK, SSMW);
  CVT(ssm_p1, dtr1_bf, MROWS, DTRANK, SSMW);
  gemm_wmma_n64<<<dim3(DINNER / 64, MROWS / 128), blk, 0, stream>>>(
      dtr0_bf, w_dt0_bf, delta0, MROWS, DINNER, DTRANK, DTRANK, DTRANK, DINNER);
  gemm_wmma_n64<<<dim3(DINNER / 64, MROWS / 128), blk, 0, stream>>>(
      dtr1_bf, w_dt1_bf, delta1, MROWS, DINNER, DTRANK, DTRANK, DTRANK, DINNER);
  // 8) depthwise conv + SiLU (h2t causal, t2h anti-causal in natural order)
  conv_silu<<<cdiv(MROWS * DINNER, TB), TB, 0, stream>>>(cam_xg, conv_w_h2t, conv_b_h2t, xact0, 0);
  conv_silu<<<cdiv(MROWS * DINNER, TB), TB, 0, stream>>>(cam_xg, conv_w_t2h, conv_b_t2h, xact1, 1);
  // 9) chunked selective scans (8x parallel over L)
  const int NPART = BB * DINNER * NCH;
  scan_partial<<<cdiv(NPART, TB), TB, 0, stream>>>(delta0, ssm_p0, xact0, A_log_h2t, dt_b_h2t, P0, S0, 0);
  scan_partial<<<cdiv(NPART, TB), TB, 0, stream>>>(delta1, ssm_p1, xact1, A_log_t2h, dt_b_t2h, P1, S1, 1);
  scan_carry<<<cdiv(BB * DINNER, TB), TB, 0, stream>>>(P0, S0, H0);
  scan_carry<<<cdiv(BB * DINNER, TB), TB, 0, stream>>>(P1, S1, H1);
  scan_apply<<<cdiv(NPART, TB), TB, 0, stream>>>(delta0, ssm_p0, xact0, cam_xz, A_log_h2t, D_h2t, dt_b_h2t, H0, y0, 0);
  scan_apply<<<cdiv(NPART, TB), TB, 0, stream>>>(delta1, ssm_p1, xact1, cam_xz, A_log_t2h, D_t2h, dt_b_t2h, H1, y1, 1);
  // 10) y_sum -> bf16, out = y_sum @ out_proj^T   [4096 x 512, K=1024]
  combine_bf16<<<cdiv(MROWS * DINNER, TB), TB, 0, stream>>>(y0, y1, ysum_bf, MROWS * DINNER);
  gemm_wmma_n64<<<dim3(DMODEL / 64, MROWS / 128), blk, 0, stream>>>(
      ysum_bf, w_out_bf, out, MROWS, DMODEL, DINNER, DINNER, DINNER, DMODEL);
#undef CVT
}
